// GCN5_83227876262528
// MI455X (gfx1250) — matrix operand order repaired and checked by reading, not compiled, not run on hardware
//
#include <hip/hip_runtime.h>
#include <hip/hip_bf16.h>

// ---------------------------------------------------------------------------
// GAT (3 blocks x 2 GATConv, H=3) for MI455X / gfx1250.
// GEMMs via v_wmma_f32_16x16x32_bf16 (fp32 accum), 4 N-tiles per wave with
// branch-free fragment loads; edge softmax/aggregation via fp32 global
// atomics (L2-resident); BN/activations fused elementwise.
// ---------------------------------------------------------------------------

typedef __attribute__((ext_vector_type(16))) __bf16 v16bf;
typedef __attribute__((ext_vector_type(8)))  float  v8f;

#define BN_EPS 1e-5f

__device__ __forceinline__ float leakyf(float v, float s) { return v > 0.f ? v : s * v; }

__device__ __forceinline__ void atomicMaxF(float* addr, float value) {
  // monotonic float-max via int/uint atomics (init must be -inf)
  if (value >= 0.f) atomicMax((int*)addr, __float_as_int(value));
  else              atomicMin((unsigned int*)addr, __float_as_uint(value));
}

// --------------------------- weight packing --------------------------------
// Pack W[K, Ncols] fp32 -> bf16 B-fragments. Element i of lane l for tile
// (tn, kk): col = tn*16 + (l&15), k = kk*32 + (l>>4)*8 + (i<8 ? i : 16 + (i-8))   (ISA 7.12.2)
__global__ void gat_pack_w(const float* __restrict__ W, __bf16* __restrict__ Bp,
                           int K, int Ncols, int nTiles, int kIters) {
  int idx = blockIdx.x * blockDim.x + threadIdx.x;
  int total = nTiles * kIters * 32;
  if (idx >= total) return;
  int lane = idx & 31;
  int kk   = (idx >> 5) % kIters;
  int tn   = (idx >> 5) / kIters;
  int col  = tn * 16 + (lane & 15);
  int kbase = kk * 32 + (lane >> 4) * 8;
  __bf16* out = Bp + (size_t)idx * 16;
#pragma unroll
  for (int i = 0; i < 16; ++i) {
    int k = kbase + (i < 8 ? i : i + 8);
    float v = (k < K && col < Ncols) ? W[(size_t)k * Ncols + col] : 0.f;
    out[i] = (__bf16)v;
  }
}

// ------------------------------ WMMA GEMM ----------------------------------
// C[M, Ncols] = A[M, K] x W ; one 16x64 output strip (4 N-tiles) per wave.
// A fragment (16-bit A 16x32 layout): lane l holds row tm*16+(l&15);
// element i -> k = k0 + (l>>4)*8 + (i<8 ? i : 16 + (i-8)).
__global__ void gat_wmma_gemm(const float* __restrict__ A, const __bf16* __restrict__ Bp,
                              float* __restrict__ C, int M, int Ncols, int K,
                              int nTiles, int kIters) {
  int nGroups = (nTiles + 3) >> 2;
  int wid  = blockIdx.x * (blockDim.x >> 5) + (threadIdx.x >> 5);
  int lane = threadIdx.x & 31;
  int mTiles = (M + 15) >> 4;
  if (wid >= mTiles * nGroups) return;
  int tm  = wid / nGroups;
  int tn0 = (wid % nGroups) * 4;
  int row = tm * 16 + (lane & 15);
  int kh  = lane >> 4;
  // clamp row: OOB rows compute garbage that is never stored (branch-free)
  int rowc = row < M ? row : (M - 1);
  const float* arow = A + (size_t)rowc * K;
  // clamp tile ids: last (partial) group recomputes a valid tile, store-guarded
  int tnc0 = (tn0 + 0) < nTiles ? (tn0 + 0) : (nTiles - 1);
  int tnc1 = (tn0 + 1) < nTiles ? (tn0 + 1) : (nTiles - 1);
  int tnc2 = (tn0 + 2) < nTiles ? (tn0 + 2) : (nTiles - 1);
  int tnc3 = (tn0 + 3) < nTiles ? (tn0 + 3) : (nTiles - 1);
  v8f c0 = {}, c1 = {}, c2 = {}, c3 = {};
  int kFull = K >> 5;  // full, unconditional iterations (K%32==0 for all but A1)
  for (int kk = 0; kk < kFull; ++kk) {
    int k0 = kk * 32 + kh * 8;
    // 32B-aligned (K%32==0 in this path): two float4 pairs, no bounds checks
    const float4* p = (const float4*)(arow + k0);
    float4 f0 = p[0], f1 = p[1];
    const float4* q = (const float4*)(arow + k0 + 16);
    float4 f2 = q[0], f3 = q[1];
    v16bf a;
    a[0]  = (__bf16)f0.x; a[1]  = (__bf16)f0.y; a[2]  = (__bf16)f0.z; a[3]  = (__bf16)f0.w;
    a[4]  = (__bf16)f1.x; a[5]  = (__bf16)f1.y; a[6]  = (__bf16)f1.z; a[7]  = (__bf16)f1.w;
    a[8]  = (__bf16)f2.x; a[9]  = (__bf16)f2.y; a[10] = (__bf16)f2.z; a[11] = (__bf16)f2.w;
    a[12] = (__bf16)f3.x; a[13] = (__bf16)f3.y; a[14] = (__bf16)f3.z; a[15] = (__bf16)f3.w;
    if (kk + 1 < kFull) __builtin_prefetch(arow + k0 + 32, 0, 0);
    v16bf b0 = *(const v16bf*)(Bp + (((size_t)(tnc0 * kIters + kk)) * 32 + lane) * 16);
    v16bf b1 = *(const v16bf*)(Bp + (((size_t)(tnc1 * kIters + kk)) * 32 + lane) * 16);
    v16bf b2 = *(const v16bf*)(Bp + (((size_t)(tnc2 * kIters + kk)) * 32 + lane) * 16);
    v16bf b3 = *(const v16bf*)(Bp + (((size_t)(tnc3 * kIters + kk)) * 32 + lane) * 16);
    c0 = __builtin_amdgcn_wmma_f32_16x16x32_bf16(false, a, false, b0, (short)0, c0, false, false);
    c1 = __builtin_amdgcn_wmma_f32_16x16x32_bf16(false, a, false, b1, (short)0, c1, false, false);
    c2 = __builtin_amdgcn_wmma_f32_16x16x32_bf16(false, a, false, b2, (short)0, c2, false, false);
    c3 = __builtin_amdgcn_wmma_f32_16x16x32_bf16(false, a, false, b3, (short)0, c3, false, false);
  }
  if (K & 31) {  // tail (only the K=3 first layer): clamp indices, cndmask not branches
    int k0 = kFull * 32;
    v16bf a;
#pragma unroll
    for (int i = 0; i < 8; ++i) {
      int ka = k0 + kh * 8 + i;
      int kb = ka + 16;
      int kac = ka < K ? ka : 0;
      int kbc = kb < K ? kb : 0;
      float va = arow[kac];
      float vb = arow[kbc];
      a[i]     = (ka < K) ? (__bf16)va : (__bf16)0.f;
      a[i + 8] = (kb < K) ? (__bf16)vb : (__bf16)0.f;
    }
    int kk = kFull;
    v16bf b0 = *(const v16bf*)(Bp + (((size_t)(tnc0 * kIters + kk)) * 32 + lane) * 16);
    v16bf b1 = *(const v16bf*)(Bp + (((size_t)(tnc1 * kIters + kk)) * 32 + lane) * 16);
    v16bf b2 = *(const v16bf*)(Bp + (((size_t)(tnc2 * kIters + kk)) * 32 + lane) * 16);
    v16bf b3 = *(const v16bf*)(Bp + (((size_t)(tnc3 * kIters + kk)) * 32 + lane) * 16);
    c0 = __builtin_amdgcn_wmma_f32_16x16x32_bf16(false, a, false, b0, (short)0, c0, false, false);
    c1 = __builtin_amdgcn_wmma_f32_16x16x32_bf16(false, a, false, b1, (short)0, c1, false, false);
    c2 = __builtin_amdgcn_wmma_f32_16x16x32_bf16(false, a, false, b2, (short)0, c2, false, false);
    c3 = __builtin_amdgcn_wmma_f32_16x16x32_bf16(false, a, false, b3, (short)0, c3, false, false);
  }
  // C/D layout: lanes 0-15 -> M = tm*16 + j, lanes 16-31 -> M = tm*16 + 8 + j
  int r0 = tm * 16 + kh * 8;
  int cl = lane & 15;
  v8f accs[4] = {c0, c1, c2, c3};
#pragma unroll
  for (int t = 0; t < 4; ++t) {
    int tnt = tn0 + t;
    int col = tnt * 16 + cl;
    if (tnt < nTiles && col < Ncols) {
#pragma unroll
      for (int j = 0; j < 8; ++j)
        if (r0 + j < M) C[(size_t)(r0 + j) * Ncols + col] = accs[t][j];
    }
  }
}

// ----------------------- attention scores per node -------------------------
__global__ void gat_scores(const float* __restrict__ h, const float* __restrict__ a_src,
                           const float* __restrict__ a_dst, float* __restrict__ es,
                           float* __restrict__ ed, int Nn, int H, int D) {
  int idx = blockIdx.x * blockDim.x + threadIdx.x;
  if (idx >= Nn * H) return;
  int head = idx % H;
  int node = idx / H;
  const float* hp = h + ((size_t)node * H + head) * D;
  const float* as = a_src + (size_t)head * D;
  const float* ad = a_dst + (size_t)head * D;
  float s = 0.f, d = 0.f;
  for (int i = 0; i < D; ++i) { float v = hp[i]; s += v * as[i]; d += v * ad[i]; }
  es[idx] = s;
  ed[idx] = d;
}

__global__ void gat_fill(float* __restrict__ p, float v, size_t n) {
  size_t i = (size_t)blockIdx.x * blockDim.x + threadIdx.x;
  if (i < n) p[i] = v;
}

// ----------------------------- edge kernels --------------------------------
__device__ __forceinline__ void edge_sd(const int* __restrict__ ei, int E, size_t e,
                                        int& s, int& d) {
  if (e < (size_t)E) { s = ei[e]; d = ei[(size_t)E + e]; }
  else               { s = d = (int)(e - (size_t)E); }   // self loop
}

__global__ void gat_edge_max(const int* __restrict__ ei, int E, int Nn, int H,
                             const float* __restrict__ es, const float* __restrict__ ed,
                             float* __restrict__ mx) {
  size_t idx = (size_t)blockIdx.x * blockDim.x + threadIdx.x;
  size_t total = (size_t)(E + Nn) * H;
  if (idx >= total) return;
  int hh = (int)(idx % H);
  size_t e = idx / H;
  int s, d; edge_sd(ei, E, e, s, d);
  float v = leakyf(es[(size_t)s * H + hh] + ed[(size_t)d * H + hh], 0.2f);
  atomicMaxF(&mx[(size_t)d * H + hh], v);
}

__global__ void gat_edge_expsum(const int* __restrict__ ei, int E, int Nn, int H,
                                const float* __restrict__ es, const float* __restrict__ ed,
                                const float* __restrict__ mx, float* __restrict__ den,
                                float* __restrict__ exb) {
  size_t idx = (size_t)blockIdx.x * blockDim.x + threadIdx.x;
  size_t total = (size_t)(E + Nn) * H;
  if (idx >= total) return;
  int hh = (int)(idx % H);
  size_t e = idx / H;
  int s, d; edge_sd(ei, E, e, s, d);
  float v  = leakyf(es[(size_t)s * H + hh] + ed[(size_t)d * H + hh], 0.2f);
  float xv = __expf(v - mx[(size_t)d * H + hh]);
  exb[idx] = xv;
  atomicAdd(&den[(size_t)d * H + hh], xv);
}

__global__ void gat_edge_accum(const int* __restrict__ ei, int E, int Nn, int H, int D,
                               int chunks, const float* __restrict__ h,
                               const float* __restrict__ exb, const float* __restrict__ den,
                               float* __restrict__ acc) {
  size_t idx = (size_t)blockIdx.x * blockDim.x + threadIdx.x;
  size_t total = (size_t)(E + Nn) * H * chunks;
  if (idx >= total) return;
  int c = (int)(idx % chunks);
  size_t t = idx / chunks;
  int hh = (int)(t % H);
  size_t e = t / H;
  int s, d; edge_sd(ei, E, e, s, d);
  float alpha = exb[e * H + hh] / den[(size_t)d * H + hh];
  const float* hs = h   + ((size_t)s * H + hh) * D;
  float*       ap = acc + ((size_t)d * H + hh) * D;
  int d0 = c * 4;
#pragma unroll
  for (int i = 0; i < 4; ++i) {
    int dd = d0 + i;
    if (dd < D) atomicAdd(&ap[dd], hs[dd] * alpha);
  }
}

// ------------------------------ finalizers ---------------------------------
__global__ void gat_fin_concat(float* __restrict__ acc, const float* __restrict__ bias,
                               size_t total, int HD, int doRelu) {
  size_t idx = (size_t)blockIdx.x * blockDim.x + threadIdx.x;
  if (idx >= total) return;
  float v = acc[idx] + bias[idx % HD];
  acc[idx] = doRelu ? fmaxf(v, 0.f) : v;
}

__global__ void gat_fin_mean(const float* __restrict__ acc, const float* __restrict__ bias,
                             float* __restrict__ out, int Nn, int H, int D, int sig) {
  size_t idx = (size_t)blockIdx.x * blockDim.x + threadIdx.x;
  if (idx >= (size_t)Nn * D) return;
  int d = (int)(idx % D);
  size_t n = idx / D;
  float s = 0.f;
  for (int h = 0; h < H; ++h) s += acc[(n * H + h) * D + d];
  s = s / (float)H + bias[d];
  if (sig) s = 1.f / (1.f + __expf(-s));
  out[idx] = s;
}

// ------------------------------ batch norm ---------------------------------
__global__ void gat_bn_stats(const float* __restrict__ x, float* __restrict__ sum,
                             float* __restrict__ sq, int Nn, int F) {
  // blockDim.x must be a multiple of F (F in {64,256}); register accumulation,
  // one pair of global atomics per thread at the end.
  int f    = threadIdx.x % F;
  int sub  = threadIdx.x / F;
  int nsub = blockDim.x / F;
  float s = 0.f, q = 0.f;
  for (int r = blockIdx.x * nsub + sub; r < Nn; r += gridDim.x * nsub) {
    float v = x[(size_t)r * F + f];
    s += v; q += v * v;
  }
  atomicAdd(&sum[f], s);
  atomicAdd(&sq[f], q);
}

__global__ void gat_bn_apply(float* __restrict__ x, const float* __restrict__ sum,
                             const float* __restrict__ sq, const float* __restrict__ g,
                             const float* __restrict__ b, int Nn, int F) {
  size_t idx = (size_t)blockIdx.x * blockDim.x + threadIdx.x;
  if (idx >= (size_t)Nn * F) return;
  int f = (int)(idx % F);
  float inv = 1.f / (float)Nn;
  float mu  = sum[f] * inv;
  float var = sq[f] * inv - mu * mu;
  float v = g[f] * (x[idx] - mu) * rsqrtf(var + BN_EPS) + b[f];
  x[idx] = leakyf(v, 0.1f);
}

// ------------------------------- host side ---------------------------------
extern "C" void kernel_launch(void* const* d_in, const int* in_sizes, int n_in,
                              void* d_out, int out_size, void* d_ws, size_t ws_size,
                              hipStream_t stream) {
  (void)n_in; (void)out_size; (void)ws_size;
  const float* x  = (const float*)d_in[0];
  const int*   ei = (const int*)d_in[1];
  const int N = in_sizes[0] / 3;
  const int E = in_sizes[1] / 2;
  const int H = 3;
  float* out = (float*)d_out;

  // workspace layout (two 307MB ping-pong feature buffers + small scratch)
  float* ws = (float*)d_ws;
  const size_t big = (size_t)N * 1536;
  float* P     = ws;
  float* Q     = P + big;
  float* es    = Q + big;
  float* ed    = es + (size_t)N * H;
  float* mx    = ed + (size_t)N * H;
  float* den   = mx + (size_t)N * H;
  float* exb   = den + (size_t)N * H;
  float* bnsum = exb + (size_t)(E + N) * H;
  float* bnsq  = bnsum + 256;
  __bf16* bp   = (__bf16*)(bnsq + 256);   // packed weights (max ~1.2 MB)

  const int T = 256;
  auto blocks = [](size_t n, int t) { return (unsigned)((n + (size_t)t - 1) / t); };
  const size_t EH = (size_t)(E + N) * H;

  auto conv = [&](const float* in, int K, int D,
                  const float* W, const float* asp, const float* adp,
                  float* h, float* acc) {
    int Ncols  = H * D;
    int nTiles = (Ncols + 15) / 16;
    int kIters = (K + 31) / 32;
    int ptot   = nTiles * kIters * 32;
    gat_pack_w<<<blocks((size_t)ptot, T), T, 0, stream>>>(W, bp, K, Ncols, nTiles, kIters);
    int nGroups = (nTiles + 3) / 4;
    size_t waves = (size_t)((N + 15) / 16) * nGroups;
    gat_wmma_gemm<<<blocks(waves * 32, T), T, 0, stream>>>(in, bp, h, N, Ncols, K,
                                                           nTiles, kIters);
    gat_scores<<<blocks((size_t)N * H, T), T, 0, stream>>>(h, asp, adp, es, ed, N, H, D);
    gat_fill<<<blocks((size_t)N * H, T), T, 0, stream>>>(mx, -INFINITY, (size_t)N * H);
    gat_fill<<<blocks((size_t)N * H, T), T, 0, stream>>>(den, 0.f, (size_t)N * H);
    gat_fill<<<blocks((size_t)N * Ncols, T), T, 0, stream>>>(acc, 0.f, (size_t)N * Ncols);
    gat_edge_max<<<blocks(EH, T), T, 0, stream>>>(ei, E, N, H, es, ed, mx);
    gat_edge_expsum<<<blocks(EH, T), T, 0, stream>>>(ei, E, N, H, es, ed, mx, den, exb);
    int chunks = (D + 3) / 4;
    gat_edge_accum<<<blocks(EH * chunks, T), T, 0, stream>>>(ei, E, N, H, D, chunks,
                                                             h, exb, den, acc);
  };

  auto bn = [&](float* xf, int F, const float* g, const float* b) {
    gat_fill<<<1, 256, 0, stream>>>(bnsum, 0.f, 256);
    gat_fill<<<1, 256, 0, stream>>>(bnsq, 0.f, 256);
    gat_bn_stats<<<256, 256, 0, stream>>>(xf, bnsum, bnsq, N, F);
    gat_bn_apply<<<blocks((size_t)N * F, T), T, 0, stream>>>(xf, bnsum, bnsq, g, b, N, F);
  };

#define PF(i) ((const float*)d_in[i])
  // Block A
  conv(x, 3, 64, PF(2), PF(3), PF(4), P, Q);                       // A1: h=P, acc=Q
  gat_fin_concat<<<blocks((size_t)N * 192, T), T, 0, stream>>>(Q, PF(5), (size_t)N * 192,
                                                               192, 1);
  conv(Q, 192, 64, PF(6), PF(7), PF(8), P, Q);                     // A2: h=P, acc=Q
  gat_fin_mean<<<blocks((size_t)N * 64, T), T, 0, stream>>>(Q, PF(9), P, N, H, 64, 0);
  bn(P, 64, PF(26), PF(27));                                       // bn1 + leaky(0.1)
  // Block B
  conv(P, 64, 256, PF(10), PF(11), PF(12), Q, P);                  // B1: h=Q, acc=P
  gat_fin_concat<<<blocks((size_t)N * 768, T), T, 0, stream>>>(P, PF(13), (size_t)N * 768,
                                                               768, 1);
  conv(P, 768, 256, PF(14), PF(15), PF(16), Q, P);                 // B2: h=Q, acc=P
  gat_fin_mean<<<blocks((size_t)N * 256, T), T, 0, stream>>>(P, PF(17), Q, N, H, 256, 0);
  bn(Q, 256, PF(28), PF(29));                                      // bn2 + leaky(0.1)
  // Block C
  conv(Q, 256, 512, PF(18), PF(19), PF(20), P, Q);                 // C1: h=P, acc=Q
  gat_fin_concat<<<blocks((size_t)N * 1536, T), T, 0, stream>>>(Q, PF(21),
                                                                (size_t)N * 1536, 1536, 1);
  conv(Q, 1536, 1, PF(22), PF(23), PF(24), P, Q);                  // C2: h=P, acc=Q
  gat_fin_mean<<<blocks((size_t)N, T), T, 0, stream>>>(Q, PF(25), out, N, H, 1, 1);
#undef PF
}
